// ReactionCodebook_50714973831818
// MI455X (gfx1250) — matrix operand order repaired and checked by reading, not compiled, not run on hardware
//
#include <hip/hip_runtime.h>
#include <stdint.h>

// ---------------------------------------------------------------------------
// VQ codebook: fused (z.W^T GEMM + argmin) with CDNA5 WMMA bf16x3, async-LDS
// double-buffered W staging, 320KB WGP LDS.
// Outputs (flat in d_out): z_q [B*D] f32, indices [B] (as f32), vq_loss [1].
// ---------------------------------------------------------------------------

typedef __attribute__((ext_vector_type(16))) __bf16 v16bf;
typedef __attribute__((ext_vector_type(8)))  float  v8f;
typedef int v4i __attribute__((vector_size(16)));   // matches async builtin proto

#define B_ROWS   32768
#define D_DIM    512
#define K_CB     8192
#define BM       128                         // z rows per block (main kernel)
#define NT       16                          // codebook entries per iteration
#define NITER    (K_CB / NT)                 // 512
#define LDS_ZBYTES (BM * D_DIM * 2 * 2)      // 262144 B: z hi+lo bf16 fragments
#define LDS_WSTAGE (NT * D_DIM * 2 * 2)      // 32768 B per W stage buffer
#define LDS_TOTAL  (LDS_ZBYTES + 2 * LDS_WSTAGE)  // 327680 = 320 KB WGP LDS

// ---- CDNA5 async global->LDS copy (ASYNCcnt), with sync fallback ----------
__device__ __forceinline__ void async_copy_16B(const void* g, void* l) {
#if __has_builtin(__builtin_amdgcn_global_load_async_to_lds_b128)
  // Probe-revealed prototype: (AS1 v4i*, AS3 v4i*, imm offset, cpol)
  __builtin_amdgcn_global_load_async_to_lds_b128(
      (__attribute__((address_space(1))) v4i*)g,
      (__attribute__((address_space(3))) v4i*)l, 0, 0);
#else
  *(uint4*)l = *(const uint4*)g;
#endif
}

__device__ __forceinline__ void wait_async0() {
#if __has_builtin(__builtin_amdgcn_s_wait_asynccnt)
  __builtin_amdgcn_s_wait_asynccnt(0);
#else
  asm volatile("s_wait_asynccnt 0" ::: "memory");
#endif
}

// Fragment layout in LDS (per 16x32 bf16 operand fragment, 1024 B):
//   bytes [0,512):   lane*16 -> halves h=0..7  of that lane's v16bf
//   bytes [512,1024): lane*16 -> halves h=8..15
// A full ds_load_b128 across 32 lanes covers a contiguous 512B block
// -> conflict-free across all 64 LDS banks.
__device__ __forceinline__ v16bf load_frag(const char* p, int lane) {
  union { uint4 u[2]; v16bf v; } t;
  t.u[0] = *(const uint4*)(p + lane * 16);
  t.u[1] = *(const uint4*)(p + 512 + lane * 16);
  return t.v;
}

// ---------------------------------------------------------------------------
// Kernel 1: split W into bf16 hi/lo and compute ||w||^2 per codebook row.
// ---------------------------------------------------------------------------
__global__ __launch_bounds__(128)
void vq_prepw_kernel(const float* __restrict__ W, __bf16* __restrict__ w_hi,
                     __bf16* __restrict__ w_lo, float* __restrict__ wnorm) {
  const int k = blockIdx.x;
  const float* row = W + (size_t)k * D_DIM;
  float s = 0.f;
  for (int c = threadIdx.x; c < D_DIM; c += 128) {
    float x = row[c];
    __bf16 h = (__bf16)x;
    w_hi[(size_t)k * D_DIM + c] = h;
    w_lo[(size_t)k * D_DIM + c] = (__bf16)(x - (float)h);
    s += x * x;
  }
  for (int off = 16; off >= 1; off >>= 1) s += __shfl_down(s, off, 32);
  __shared__ float red[4];
  if ((threadIdx.x & 31) == 0) red[threadIdx.x >> 5] = s;
  __syncthreads();
  if (threadIdx.x == 0) wnorm[k] = red[0] + red[1] + red[2] + red[3];
}

// ---------------------------------------------------------------------------
// Kernel 2: fused distance GEMM + argmin.
// Block = 256 threads = 8 waves; wave w owns z-rows [w*16, w*16+16).
// z tile (128x512, bf16 hi/lo) resident in LDS; W streamed 16 rows/iter via
// double-buffered async LDS stage. dist = ||w||^2 - 2*(z.w)  (||z||^2 const).
// ---------------------------------------------------------------------------
__global__ __launch_bounds__(256)
void vq_main_kernel(const float* __restrict__ z,
                    const __bf16* __restrict__ w_hi,
                    const __bf16* __restrict__ w_lo,
                    const float* __restrict__ wnorm,
                    float* __restrict__ outIdx) {
  extern __shared__ char smem[];
  const int tid  = threadIdx.x;
  const int lane = tid & 31;
  const int wave = tid >> 5;
  const int blockRow = blockIdx.x * BM;

  // ---- Phase 1: load block's z rows, split to bf16 hi/lo A-fragments -----
  // A-fragment (16x32, 16-bit) lane/half mapping per ISA 7.12.2:
  //   lane<16: M=lane,    k = (h<8 ? h : h+8)
  //   lane>=16:M=lane-16, k = (h<8 ? h : h+8) + 8
  for (int idx = tid; idx < BM * D_DIM; idx += 256) {
    int h  = idx & 15;
    int l  = (idx >> 4) & 31;
    int kk = (idx >> 9) & 15;
    int m  = idx >> 13;
    int row = m * 16 + (l & 15);
    int k = kk * 32 + ((h < 8) ? h : h + 8) + ((l >= 16) ? 8 : 0);
    float x = z[(size_t)(blockRow + row) * D_DIM + k];
    __bf16 hi = (__bf16)x;
    __bf16 lo = (__bf16)(x - (float)hi);
    char* fh = smem + ((m * 16 + kk) * 2 + 0) * 1024
             + ((h < 8) ? 0 : 512) + l * 16 + (h & 7) * 2;
    *(__bf16*)fh = hi;
    *(__bf16*)(fh + 1024) = lo;   // lo fragment follows hi fragment
  }

  // ---- W stage: copy 16 codebook rows (hi+lo) into B-fragment layout -----
  // B-fragment (32x16): lane l -> N=l%16, k = (h<8 ? h : h+8) + (l>=16 ? 8:0).
  // Each 16B chunk is 8 contiguous halves of one W row.
  auto stageW = [&](int nc, int buf) {
    char* wb = smem + LDS_ZBYTES + buf * LDS_WSTAGE;
#pragma unroll
    for (int c8 = 0; c8 < 8; ++c8) {
      int c    = c8 * 256 + tid;        // 2048 chunks of 16B = 32KB
      int part = c & 1;                 // halves 0..7 vs 8..15
      int l    = (c >> 1) & 31;
      int ph   = (c >> 6) & 1;          // 0=hi, 1=lo
      int kk   = c >> 7;                // 0..15
      const __bf16* base = ph ? w_lo : w_hi;
      const __bf16* src = base + (size_t)(nc * NT + (l & 15)) * D_DIM
                        + kk * 32 + ((l >= 16) ? 8 : 0) + part * 16;
      char* dst = wb + (kk * 2 + ph) * 1024 + part * 512 + l * 16;
      async_copy_16B(src, dst);
    }
  };

  __syncthreads();
  stageW(0, 0);
  wait_async0();
  __syncthreads();

  float bestD[8], bestI[8];
#pragma unroll
  for (int r = 0; r < 8; ++r) { bestD[r] = 3.4e38f; bestI[r] = 0.f; }

  const char* zb = smem + (wave * 16 * 2) * 1024;   // this wave's A fragments

  for (int nc = 0; nc < NITER; ++nc) {
    if (nc + 1 < NITER) stageW(nc + 1, (nc + 1) & 1);   // prefetch next buf

    const char* wb = smem + LDS_ZBYTES + (nc & 1) * LDS_WSTAGE;
    v8f accHH = {}, accHL = {}, accLH = {};     // 3 independent WMMA chains
#pragma unroll
    for (int kk = 0; kk < 16; ++kk) {
      v16bf ah = load_frag(zb + (kk * 2 + 0) * 1024, lane);
      v16bf al = load_frag(zb + (kk * 2 + 1) * 1024, lane);
      v16bf bh = load_frag(wb + (kk * 2 + 0) * 1024, lane);
      v16bf bl = load_frag(wb + (kk * 2 + 1) * 1024, lane);
      accHH = __builtin_amdgcn_wmma_f32_16x16x32_bf16(false, ah, false, bh,
                                                      (short)0, accHH, false, false);
      accHL = __builtin_amdgcn_wmma_f32_16x16x32_bf16(false, ah, false, bl,
                                                      (short)0, accHL, false, false);
      accLH = __builtin_amdgcn_wmma_f32_16x16x32_bf16(false, al, false, bh,
                                                      (short)0, accLH, false, false);
    }
    // C/D layout: VGPR r, lane l -> M = r + (l>=16?8:0), N = l%16
    float wn = wnorm[nc * NT + (lane & 15)];
    float ni = (float)(nc * NT + (lane & 15));
#pragma unroll
    for (int r = 0; r < 8; ++r) {
      float dist = wn - 2.0f * (accHH[r] + accHL[r] + accLH[r]);
      if (dist < bestD[r]) { bestD[r] = dist; bestI[r] = ni; }
    }
    wait_async0();       // next buffer fully staged
    __syncthreads();     // everyone done reading current buffer
  }

  // ---- argmin reduction across the 16 lanes sharing each output row ------
#pragma unroll
  for (int r = 0; r < 8; ++r) {
    float d = bestD[r], i = bestI[r];
#pragma unroll
    for (int mk = 8; mk >= 1; mk >>= 1) {   // stays within 16-lane halves
      float d2 = __shfl_xor(d, mk, 32);
      float i2 = __shfl_xor(i, mk, 32);
      if (d2 < d || (d2 == d && i2 < i)) { d = d2; i = i2; }
    }
    if ((lane & 15) == 0) {
      int row = blockRow + wave * 16 + r + ((lane >= 16) ? 8 : 0);
      outIdx[row] = i;
    }
  }
}

// ---------------------------------------------------------------------------
// Kernel 3: gather z_q = W[idx] (straight-through output) + per-block MSE.
// ---------------------------------------------------------------------------
__global__ __launch_bounds__(256)
void vq_gather_kernel(const float* __restrict__ z, const float* __restrict__ W,
                      const float* __restrict__ idxF, float* __restrict__ zq,
                      float* __restrict__ partial) {
  float s = 0.f;
  for (int r = 0; r < 64; ++r) {
    int row = blockIdx.x * 64 + r;
    int k = (int)idxF[row];
    const float* wr = W + (size_t)k * D_DIM;
    const float* zr = z + (size_t)row * D_DIM;
    float* qr = zq + (size_t)row * D_DIM;
    for (int c = threadIdx.x; c < D_DIM; c += 256) {
      float wv = wr[c];
      float df = wv - zr[c];
      qr[c] = wv;
      s += df * df;
    }
  }
  __shared__ float red[8];
  for (int off = 16; off >= 1; off >>= 1) s += __shfl_down(s, off, 32);
  if ((threadIdx.x & 31) == 0) red[threadIdx.x >> 5] = s;
  __syncthreads();
  if (threadIdx.x == 0) {
    float t = 0.f;
    for (int w = 0; w < 8; ++w) t += red[w];
    partial[blockIdx.x] = t;
  }
}

// ---------------------------------------------------------------------------
// Kernel 4: deterministic final loss: vq_loss = (1 + 0.25) * MSE.
// ---------------------------------------------------------------------------
__global__ __launch_bounds__(256)
void vq_loss_kernel(const float* __restrict__ partial, float* __restrict__ lossOut) {
  float s = partial[threadIdx.x] + partial[threadIdx.x + 256];
  __shared__ float red[8];
  for (int off = 16; off >= 1; off >>= 1) s += __shfl_down(s, off, 32);
  if ((threadIdx.x & 31) == 0) red[threadIdx.x >> 5] = s;
  __syncthreads();
  if (threadIdx.x == 0) {
    float t = 0.f;
    for (int w = 0; w < 8; ++w) t += red[w];
    lossOut[0] = 1.25f * t / (float)((size_t)B_ROWS * D_DIM);
  }
}

// ---------------------------------------------------------------------------
extern "C" void kernel_launch(void* const* d_in, const int* in_sizes, int n_in,
                              void* d_out, int out_size, void* d_ws, size_t ws_size,
                              hipStream_t stream) {
  (void)in_sizes; (void)n_in; (void)out_size; (void)ws_size;
  const float* z = (const float*)d_in[0];   // [32768, 512] f32
  const float* W = (const float*)d_in[1];   // [8192, 512] f32

  float* out = (float*)d_out;
  float* outIdx  = out + (size_t)B_ROWS * D_DIM;        // [B] indices as f32
  float* outLoss = outIdx + B_ROWS;                     // [1] vq_loss

  // Workspace layout (~16.1 MB):
  char* ws = (char*)d_ws;
  __bf16* w_hi   = (__bf16*)ws;                                   // 8 MB
  __bf16* w_lo   = (__bf16*)(ws + (size_t)K_CB * D_DIM * 2);      // 8 MB
  float*  wnorm  = (float*)(ws + (size_t)K_CB * D_DIM * 4);       // 32 KB
  float*  partial = wnorm + K_CB;                                 // 2 KB

  vq_prepw_kernel<<<K_CB, 128, 0, stream>>>(W, w_hi, w_lo, wnorm);
  vq_main_kernel<<<B_ROWS / BM, 256, LDS_TOTAL, stream>>>(z, w_hi, w_lo, wnorm, outIdx);
  vq_gather_kernel<<<B_ROWS / 64, 256, 0, stream>>>(z, W, outIdx, out, partial);
  vq_loss_kernel<<<1, 256, 0, stream>>>(partial, outLoss);
}
